// CTM_part_82815559402222
// MI455X (gfx1250) — matrix operand order repaired and checked by reading, not compiled, not run on hardware
//
#include <hip/hip_runtime.h>
#include <cstdint>

typedef unsigned short u16t;
typedef __attribute__((ext_vector_type(16))) __bf16 v16bf;
typedef __attribute__((ext_vector_type(8)))  float  v8f;

#define Bn     4
#define Ntok   4096
#define CinC   256
#define CoutC  512
#define Hm     64
#define Wm     64
#define HoW    32
#define Sc     1024
#define Kconv  2304
#define INVSQRT512 0.04419417382415922f

__device__ __forceinline__ u16t f2bf(float f){
  unsigned u = __float_as_uint(f);
  u = u + 0x7FFFu + ((u >> 16) & 1u);      // round-to-nearest-even
  return (u16t)(u >> 16);
}

union Frag { unsigned u[8]; uint4 q[2]; v16bf v; };

// A fragment, 16-bit 16x32: lanes 0-15 rows M=0..15 (K 0..7,16..23),
// lanes 16-31 same rows (K 8..15,24..31). Row-major source, K contiguous.
__device__ __forceinline__ v16bf frag_a(const u16t* rowbase, int k0, int half){
  Frag f; const uint4* p = (const uint4*)(rowbase + k0 + (half ? 8 : 0));
  f.q[0] = p[0]; f.q[1] = p[2];
  return f.v;
}
// B fragment, 16-bit 32x16: lane n holds column n; lanes 0-15 K=0..7,16..23,
// lanes 16-31 K=8..15,24..31. Source is B^T row-major (row n = column n of B).
__device__ __forceinline__ v16bf frag_b(const u16t* rowbase, int k0, int half){
  Frag f; const uint4* p = (const uint4*)(rowbase + k0 + (half ? 8 : 0));
  f.q[0] = p[0]; f.q[1] = p[2];
  return f.v;
}

__device__ __forceinline__ v8f wmma_bf16(v16bf a, v16bf b, v8f c){
  return __builtin_amdgcn_wmma_f32_16x16x32_bf16(false, a, false, b, (short)0, c, false, false);
}

// ---------------- token2map scatter ----------------
__global__ void k_scatter_map(const float* __restrict__ x, const float* __restrict__ loc,
                              const int* __restrict__ idxagg, float* acc, float* cnt){
  int id = blockIdx.x * blockDim.x + threadIdx.x;
  const int total = Bn * Ntok * (CinC / 4);
  if (id >= total) return;
  int cg = id % (CinC / 4);
  int t  = id / (CinC / 4);
  int b = t >> 12;
  float lx = loc[(size_t)t * 2 + 0], ly = loc[(size_t)t * 2 + 1];
  lx = fminf(fmaxf(lx, -1.f), 1.f); ly = fminf(fmaxf(ly, -1.f), 1.f);
  int px = (int)rintf((lx + 1.f) * Wm * 0.5f - 0.5f); px = min(max(px, 0), Wm - 1);
  int py = (int)rintf((ly + 1.f) * Hm * 0.5f - 0.5f); py = min(max(py, 0), Hm - 1);
  int pix = py * Wm + px;
  int j = idxagg[t];
  const float* src = x + ((size_t)b * Ntok + j) * CinC + cg * 4;
  float* dst = acc + ((size_t)b * (Hm * Wm) + pix) * CinC + cg * 4;
  #pragma unroll
  for (int c = 0; c < 4; ++c) atomicAdd(dst + c, src[c]);
  if (cg == 0) atomicAdd(cnt + b * (Hm * Wm) + pix, 1.0f);
}

__global__ void k_map_norm(const float* __restrict__ acc, const float* __restrict__ cnt,
                           u16t* __restrict__ mapbf){
  int i = blockIdx.x * blockDim.x + threadIdx.x;
  const int total = Bn * Hm * Wm * CinC;
  if (i >= total) return;
  mapbf[i] = f2bf(acc[i] / (cnt[i >> 8] + 1e-6f));   // CinC==256
}

// dst[n*Kd+k] = bf16(src[k*Nc+n])  (transpose [K,N] -> [N,K])
__global__ void k_transpose_bf16(const float* __restrict__ src, u16t* __restrict__ dst,
                                 int Kd, int Nc){
  int id = blockIdx.x * blockDim.x + threadIdx.x;
  if (id >= Kd * Nc) return;
  int n = id / Kd, k = id - n * Kd;
  dst[(size_t)n * Kd + k] = f2bf(src[(size_t)k * Nc + n]);
}

__global__ void k_cast_bf16(const float* __restrict__ s, u16t* __restrict__ d, int n){
  int i = blockIdx.x * blockDim.x + threadIdx.x;
  if (i < n) d[i] = f2bf(s[i]);
}

// ---------------- conv 3x3 stride2 pad1 as implicit WMMA GEMM ----------------
// M = Bn*32*32 = 4096 pixels, N = 512, K = 2304; wt = conv_w^T [512][2304]
// 16x32 tile per wave: one A fragment feeds two N-tiles (halves im2col A traffic).
__global__ void k_conv_wmma(const u16t* __restrict__ map, const u16t* __restrict__ wt,
                            const float* __restrict__ bias, float* __restrict__ out){
  int gtid = blockIdx.x * blockDim.x + threadIdx.x;
  int wid = gtid >> 5, lane = gtid & 31;
  int tM = wid >> 4, tN = (wid & 15) * 32;          // 256 x 16 wave tiles
  if (tM >= 256) return;
  int half = lane >> 4, l16 = lane & 15;
  int m = tM * 16 + l16;                            // pixel row this lane loads
  int b = m >> 10, p = m & 1023, oy = p >> 5, ox = p & 31;
  const u16t* brow0 = wt + (size_t)(tN + l16) * Kconv;
  const u16t* brow1 = brow0 + (size_t)16 * Kconv;
  v8f acc0 = {}, acc1 = {};
  for (int k0 = 0; k0 < Kconv; k0 += 32){
    int tap = k0 >> 8;                              // 9 taps, 256 ci each
    int ci0 = k0 & 255;
    int ky = tap / 3, kx = tap - ky * 3;
    int iy = oy * 2 + ky - 1, ix = ox * 2 + kx - 1;
    Frag fa;
    if ((unsigned)iy < (unsigned)Hm && (unsigned)ix < (unsigned)Wm){
      const u16t* arow = map + (((size_t)(b * Hm + iy) * Wm + ix) * CinC + ci0);
      const uint4* pa = (const uint4*)(arow + (half ? 8 : 0));
      fa.q[0] = pa[0]; fa.q[1] = pa[2];
    } else {
      #pragma unroll
      for (int j = 0; j < 8; ++j) fa.u[j] = 0u;
    }
    v16bf b0 = frag_b(brow0, k0, half);
    v16bf b1 = frag_b(brow1, k0, half);
    acc0 = wmma_bf16(fa.v, b0, acc0);
    acc1 = wmma_bf16(fa.v, b1, acc1);
  }
  int n0 = tN + l16, n1 = n0 + 16;
  float bn0 = bias[n0], bn1 = bias[n1];
  #pragma unroll
  for (int r = 0; r < 8; ++r){
    int mm = tM * 16 + r + (half ? 8 : 0);
    out[(size_t)mm * CoutC + n0] = acc0[r] + bn0;
    out[(size_t)mm * CoutC + n1] = acc1[r] + bn1;
  }
}

// ---------------- generic WMMA GEMM, 32x32 tile per wave ----------------
// C[M,Nc] = A[M,Kd] * Bt[Nc,Kd]^T
__global__ void k_wmma_gemm2(const u16t* __restrict__ A, const u16t* __restrict__ Bt,
                             float* __restrict__ C, int M, int Nc, int Kd){
  int gtid = blockIdx.x * blockDim.x + threadIdx.x;
  int wid = gtid >> 5, lane = gtid & 31;
  int tilesN = Nc >> 5;
  int tM = (wid / tilesN) * 32, tN = (wid % tilesN) * 32;
  if (tM >= M) return;
  int half = lane >> 4, l16 = lane & 15;
  const u16t* arow0 = A  + (size_t)(tM + l16) * Kd;
  const u16t* arow1 = arow0 + (size_t)16 * Kd;
  const u16t* brow0 = Bt + (size_t)(tN + l16) * Kd;
  const u16t* brow1 = brow0 + (size_t)16 * Kd;
  v8f acc00 = {}, acc01 = {}, acc10 = {}, acc11 = {};
  for (int k0 = 0; k0 < Kd; k0 += 32){
    __builtin_prefetch(arow0 + k0 + 256, 0, 1);     // global_prefetch_b8
    v16bf a0 = frag_a(arow0, k0, half);
    v16bf a1 = frag_a(arow1, k0, half);
    v16bf b0 = frag_b(brow0, k0, half);
    v16bf b1 = frag_b(brow1, k0, half);
    acc00 = wmma_bf16(a0, b0, acc00);
    acc01 = wmma_bf16(a0, b1, acc01);
    acc10 = wmma_bf16(a1, b0, acc10);
    acc11 = wmma_bf16(a1, b1, acc11);
  }
  int n0 = tN + l16, n1 = n0 + 16;
  #pragma unroll
  for (int r = 0; r < 8; ++r){
    int mm0 = tM + r + (half ? 8 : 0);
    int mm1 = mm0 + 16;
    C[(size_t)mm0 * Nc + n0] = acc00[r];
    C[(size_t)mm0 * Nc + n1] = acc01[r];
    C[(size_t)mm1 * Nc + n0] = acc10[r];
    C[(size_t)mm1 * Nc + n1] = acc11[r];
  }
}

// ---------------- map2token gather + weighted scatter ----------------
__global__ void k_map2token(const float* __restrict__ conv_out, const float* __restrict__ loc,
                            const int* __restrict__ idxagg, const float* __restrict__ aggw,
                            float* mpacc, float* mpw){
  int id = blockIdx.x * blockDim.x + threadIdx.x;
  const int total = Bn * Ntok * (CoutC / 4);
  if (id >= total) return;
  int cg = id & 127;
  int t  = id >> 7;
  int b = t >> 12;
  float lx = loc[(size_t)t * 2 + 0], ly = loc[(size_t)t * 2 + 1];
  lx = fminf(fmaxf(lx, -1.f), 1.f); ly = fminf(fmaxf(ly, -1.f), 1.f);
  int px = (int)rintf((lx + 1.f) * HoW * 0.5f - 0.5f); px = min(max(px, 0), HoW - 1);
  int py = (int)rintf((ly + 1.f) * HoW * 0.5f - 0.5f); py = min(max(py, 0), HoW - 1);
  int pix = py * HoW + px;
  float w = aggw[t];
  int j = idxagg[t];
  const float* src = conv_out + ((size_t)b * (HoW * HoW) + pix) * CoutC + cg * 4;
  float* dst = mpacc + ((size_t)b * Ntok + j) * CoutC + cg * 4;
  #pragma unroll
  for (int c = 0; c < 4; ++c) atomicAdd(dst + c, src[c] * w);
  if (cg == 0) atomicAdd(mpw + b * Ntok + j, w);
}

// ---------------- combine + layernorm + conf (one 128-thread block per token) ----------------
__device__ __forceinline__ float block_sum(float v, float* red){
  int tid = threadIdx.x;
  red[tid] = v; __syncthreads();
  for (int s = 64; s > 0; s >>= 1){
    if (tid < s) red[tid] += red[tid + s];
    __syncthreads();
  }
  float r = red[0]; __syncthreads();
  return r;
}

__global__ __launch_bounds__(128) void k_token_fuse(
    const float* __restrict__ mpacc, const float* __restrict__ mpw,
    const float* __restrict__ lng, const float* __restrict__ lnb,
    const float* __restrict__ confw, const float* __restrict__ confb,
    float* __restrict__ xtok,        // in: skip-GEMM output; out: layernormed tokens
    u16t* __restrict__ xb, float* __restrict__ x2, float* __restrict__ wgt){
  __shared__ float red[128];
  int row = blockIdx.x, tid = threadIdx.x;
  size_t base = (size_t)row * CoutC + tid * 4;
  float ws = mpw[row] + 1e-6f;
  float v[4]; float s = 0.f;
  #pragma unroll
  for (int i = 0; i < 4; ++i){
    v[i] = mpacc[base + i] / ws + xtok[base + i];
    s += v[i];
  }
  float mean = block_sum(s, red) * (1.f / CoutC);
  float vs = 0.f;
  #pragma unroll
  for (int i = 0; i < 4; ++i){ v[i] -= mean; vs += v[i] * v[i]; }
  float var = block_sum(vs, red) * (1.f / CoutC);
  float rstd = rsqrtf(var + 1e-5f);
  float s2 = 0.f, scf = 0.f;
  #pragma unroll
  for (int i = 0; i < 4; ++i){
    int c = tid * 4 + i;
    float tv = v[i] * rstd * lng[c] + lnb[c];
    xtok[base + i] = tv;
    xb[base + i] = f2bf(tv);
    s2 += tv * tv;
    scf += tv * confw[c];
  }
  float x2v = block_sum(s2, red);
  float cf  = block_sum(scf, red);
  if (tid == 0){
    x2[row] = x2v;
    wgt[row] = expf(cf + confb[0]);
  }
}

// ---------------- density pass: streaming X*X^T (32x32/wave), per-row top-5 ----------------
__global__ __launch_bounds__(32) void k_density(const u16t* __restrict__ xb,
                                                const float* __restrict__ x2,
                                                float* __restrict__ density,
                                                unsigned* __restrict__ dmaxu){
  __shared__ float tile[32 * 32];
  __shared__ float x2s[32];
  int lane = threadIdx.x, wid = blockIdx.x;
  int b = wid >> 7, rb = wid & 127, row0 = rb * 32;
  int half = lane >> 4, l16 = lane & 15;
  const u16t* arow0 = xb + ((size_t)b * Ntok + row0 + l16) * CoutC;
  const u16t* arow1 = arow0 + (size_t)16 * CoutC;
  x2s[lane] = x2[b * Ntok + row0 + lane];
  __syncthreads();
  float t0 = 3.4e38f, t1 = 3.4e38f, t2 = 3.4e38f, t3 = 3.4e38f, t4 = 3.4e38f;
  float lmax = 0.f;
  for (int cg = 0; cg < Ntok / 32; ++cg){
    int col0 = cg * 32;
    const u16t* brow0 = xb + ((size_t)b * Ntok + col0 + l16) * CoutC;
    const u16t* brow1 = brow0 + (size_t)16 * CoutC;
    v8f acc00 = {}, acc01 = {}, acc10 = {}, acc11 = {};
    #pragma unroll
    for (int k0 = 0; k0 < CoutC; k0 += 32){
      v16bf a0 = frag_a(arow0, k0, half);
      v16bf a1 = frag_a(arow1, k0, half);
      v16bf b0 = frag_b(brow0, k0, half);
      v16bf b1 = frag_b(brow1, k0, half);
      acc00 = wmma_bf16(a0, b0, acc00);
      acc01 = wmma_bf16(a0, b1, acc01);
      acc10 = wmma_bf16(a1, b0, acc10);
      acc11 = wmma_bf16(a1, b1, acc11);
    }
    float x2c0 = x2[b * Ntok + col0 + l16];
    float x2c1 = x2[b * Ntok + col0 + 16 + l16];
    #pragma unroll
    for (int r = 0; r < 8; ++r){
      int mr0 = r + (half ? 8 : 0);
      int mr1 = mr0 + 16;
      float d;
      d = sqrtf(fmaxf(x2s[mr0] + x2c0 - 2.f * acc00[r], 0.f)) * INVSQRT512;
      tile[mr0 * 32 + l16] = d;      lmax = fmaxf(lmax, d);
      d = sqrtf(fmaxf(x2s[mr0] + x2c1 - 2.f * acc01[r], 0.f)) * INVSQRT512;
      tile[mr0 * 32 + 16 + l16] = d; lmax = fmaxf(lmax, d);
      d = sqrtf(fmaxf(x2s[mr1] + x2c0 - 2.f * acc10[r], 0.f)) * INVSQRT512;
      tile[mr1 * 32 + l16] = d;      lmax = fmaxf(lmax, d);
      d = sqrtf(fmaxf(x2s[mr1] + x2c1 - 2.f * acc11[r], 0.f)) * INVSQRT512;
      tile[mr1 * 32 + 16 + l16] = d; lmax = fmaxf(lmax, d);
    }
    __syncthreads();
    // every lane owns one of the 32 rows
    #pragma unroll
    for (int j = 0; j < 32; ++j){
      float v = tile[lane * 32 + j];
      if (v < t4){
        t4 = v;
        if (t4 < t3){ float tmp = t3; t3 = t4; t4 = tmp; }
        if (t3 < t2){ float tmp = t2; t2 = t3; t3 = tmp; }
        if (t2 < t1){ float tmp = t1; t1 = t2; t2 = tmp; }
        if (t1 < t0){ float tmp = t0; t0 = t1; t1 = tmp; }
      }
    }
    __syncthreads();
  }
  atomicMax(dmaxu + b, __float_as_uint(lmax));
  float msq = (t0*t0 + t1*t1 + t2*t2 + t3*t3 + t4*t4) * 0.2f;
  density[b * Ntok + row0 + lane] = expf(-msq);
}

// ---------------- parent pass: masked min dist over higher-density cols -> score ----------------
__global__ __launch_bounds__(32) void k_parent_score(const u16t* __restrict__ xb,
                                                     const float* __restrict__ x2,
                                                     const float* __restrict__ density,
                                                     const unsigned* __restrict__ dmaxu,
                                                     float* __restrict__ score){
  __shared__ float tile[32 * 32];
  __shared__ float x2s[32];
  __shared__ float dcols[32];
  int lane = threadIdx.x, wid = blockIdx.x;
  int b = wid >> 7, rb = wid & 127, row0 = rb * 32;
  int half = lane >> 4, l16 = lane & 15;
  const u16t* arow0 = xb + ((size_t)b * Ntok + row0 + l16) * CoutC;
  const u16t* arow1 = arow0 + (size_t)16 * CoutC;
  x2s[lane] = x2[b * Ntok + row0 + lane];
  __syncthreads();
  float drow = density[b * Ntok + row0 + lane];
  float minv = __uint_as_float(dmaxu[b]);
  for (int cg = 0; cg < Ntok / 32; ++cg){
    int col0 = cg * 32;
    const u16t* brow0 = xb + ((size_t)b * Ntok + col0 + l16) * CoutC;
    const u16t* brow1 = brow0 + (size_t)16 * CoutC;
    v8f acc00 = {}, acc01 = {}, acc10 = {}, acc11 = {};
    #pragma unroll
    for (int k0 = 0; k0 < CoutC; k0 += 32){
      v16bf a0 = frag_a(arow0, k0, half);
      v16bf a1 = frag_a(arow1, k0, half);
      v16bf b0 = frag_b(brow0, k0, half);
      v16bf b1 = frag_b(brow1, k0, half);
      acc00 = wmma_bf16(a0, b0, acc00);
      acc01 = wmma_bf16(a0, b1, acc01);
      acc10 = wmma_bf16(a1, b0, acc10);
      acc11 = wmma_bf16(a1, b1, acc11);
    }
    float x2c0 = x2[b * Ntok + col0 + l16];
    float x2c1 = x2[b * Ntok + col0 + 16 + l16];
    #pragma unroll
    for (int r = 0; r < 8; ++r){
      int mr0 = r + (half ? 8 : 0);
      int mr1 = mr0 + 16;
      tile[mr0 * 32 + l16]      = sqrtf(fmaxf(x2s[mr0] + x2c0 - 2.f * acc00[r], 0.f)) * INVSQRT512;
      tile[mr0 * 32 + 16 + l16] = sqrtf(fmaxf(x2s[mr0] + x2c1 - 2.f * acc01[r], 0.f)) * INVSQRT512;
      tile[mr1 * 32 + l16]      = sqrtf(fmaxf(x2s[mr1] + x2c0 - 2.f * acc10[r], 0.f)) * INVSQRT512;
      tile[mr1 * 32 + 16 + l16] = sqrtf(fmaxf(x2s[mr1] + x2c1 - 2.f * acc11[r], 0.f)) * INVSQRT512;
    }
    dcols[lane] = density[b * Ntok + col0 + lane];
    __syncthreads();
    #pragma unroll
    for (int j = 0; j < 32; ++j){
      if (dcols[j] > drow){
        float v = tile[lane * 32 + j];
        if (v < minv) minv = v;
      }
    }
    __syncthreads();
  }
  score[b * Ntok + row0 + lane] = minv * drow;
}

// ---------------- exact top-S selection by rank counting (matches jax top_k order) ------------
__global__ __launch_bounds__(1024) void k_select(const float* __restrict__ score,
                                                 int* __restrict__ index_down){
  __shared__ float sc[Ntok];
  int b = blockIdx.x, tid = threadIdx.x;
  for (int i = tid; i < Ntok; i += blockDim.x) sc[i] = score[b * Ntok + i];
  __syncthreads();
  for (int i = tid; i < Ntok; i += blockDim.x){
    float s = sc[i];
    int rank = 0;
    for (int j = 0; j < Ntok; ++j){
      float t = sc[j];
      rank += (t > s) || (t == s && j < i);
    }
    if (rank < Sc) index_down[b * Sc + rank] = i;
  }
}

__global__ void k_gather_centers(const u16t* __restrict__ xb, const float* __restrict__ x2,
                                 const int* __restrict__ idxd,
                                 u16t* __restrict__ xc, float* __restrict__ x2c){
  int id = blockIdx.x * blockDim.x + threadIdx.x;
  const int total = Bn * Sc * (CoutC / 4);
  if (id >= total) return;
  int cg = id & (CoutC / 4 - 1);
  int t  = id / (CoutC / 4);
  int b = t >> 10;
  int src = idxd[t];
  const uint2* sp = (const uint2*)(xb + ((size_t)b * Ntok + src) * CoutC) + cg;
  uint2* dp = (uint2*)(xc + (size_t)t * CoutC) + cg;
  *dp = *sp;
  if (cg == 0) x2c[t] = x2[b * Ntok + src];
}

// ---------------- assignment pass: argmin over 1024 centers per token (32x32/wave) ------------
__global__ __launch_bounds__(32) void k_assign(const u16t* __restrict__ xc,
                                               const u16t* __restrict__ xb,
                                               const float* __restrict__ x2c,
                                               const float* __restrict__ x2,
                                               int* __restrict__ idxc){
  __shared__ float tile[32 * 32];
  __shared__ float x2cs[32];
  int lane = threadIdx.x, wid = blockIdx.x;
  int b = wid >> 7, tb = wid & 127, n0 = tb * 32;
  int half = lane >> 4, l16 = lane & 15;
  const u16t* brow0 = xb + ((size_t)b * Ntok + n0 + l16) * CoutC;
  const u16t* brow1 = brow0 + (size_t)16 * CoutC;
  float x2t0 = x2[b * Ntok + n0 + l16];
  float x2t1 = x2[b * Ntok + n0 + 16 + l16];
  float minv = 3.4e38f; int amin = 0;
  for (int ct = 0; ct < Sc / 32; ++ct){
    int s0 = ct * 32;
    const u16t* arow0 = xc + ((size_t)b * Sc + s0 + l16) * CoutC;
    const u16t* arow1 = arow0 + (size_t)16 * CoutC;
    v8f acc00 = {}, acc01 = {}, acc10 = {}, acc11 = {};
    #pragma unroll
    for (int k0 = 0; k0 < CoutC; k0 += 32){
      v16bf a0 = frag_a(arow0, k0, half);
      v16bf a1 = frag_a(arow1, k0, half);
      v16bf b0 = frag_b(brow0, k0, half);
      v16bf b1 = frag_b(brow1, k0, half);
      acc00 = wmma_bf16(a0, b0, acc00);
      acc01 = wmma_bf16(a0, b1, acc01);
      acc10 = wmma_bf16(a1, b0, acc10);
      acc11 = wmma_bf16(a1, b1, acc11);
    }
    x2cs[lane] = x2c[b * Sc + s0 + lane];
    __syncthreads();
    #pragma unroll
    for (int r = 0; r < 8; ++r){
      int sr0 = r + (half ? 8 : 0);   // local center row
      int sr1 = sr0 + 16;
      tile[sr0 * 32 + l16]      = fmaxf(x2cs[sr0] + x2t0 - 2.f * acc00[r], 0.f);
      tile[sr0 * 32 + 16 + l16] = fmaxf(x2cs[sr0] + x2t1 - 2.f * acc01[r], 0.f);
      tile[sr1 * 32 + l16]      = fmaxf(x2cs[sr1] + x2t0 - 2.f * acc10[r], 0.f);
      tile[sr1 * 32 + 16 + l16] = fmaxf(x2cs[sr1] + x2t1 - 2.f * acc11[r], 0.f);
    }
    __syncthreads();
    // lane owns token column = lane; scan 32 centers (ascending keeps first min)
    #pragma unroll
    for (int sl = 0; sl < 32; ++sl){
      float v = tile[sl * 32 + lane];
      if (v < minv){ minv = v; amin = s0 + sl; }
    }
    __syncthreads();
  }
  idxc[b * Ntok + n0 + lane] = amin;
}

// ---------------- merge + finals ----------------
__global__ void k_allw(const float* __restrict__ wgt, const int* __restrict__ idxc,
                       float* __restrict__ allw){
  int i = blockIdx.x * blockDim.x + threadIdx.x;
  if (i >= Bn * Ntok) return;
  int b = i >> 12;
  atomicAdd(allw + b * Sc + idxc[i], wgt[i]);
}

__global__ void k_normw(const float* __restrict__ wgt, const float* __restrict__ allw,
                        const int* __restrict__ idxc, float* __restrict__ normw){
  int i = blockIdx.x * blockDim.x + threadIdx.x;
  if (i >= Bn * Ntok) return;
  int b = i >> 12;
  normw[i] = wgt[i] / (allw[b * Sc + idxc[i]] + 1e-6f);
}

__global__ void k_xdown(const float* __restrict__ xtok, const int* __restrict__ idxc,
                        const float* __restrict__ normw, float* __restrict__ outx){
  int id = blockIdx.x * blockDim.x + threadIdx.x;
  const int total = Bn * Ntok * (CoutC / 4);
  if (id >= total) return;
  int cg = id & 127;
  int t  = id >> 7;
  int b = t >> 12;
  int cl = idxc[t];
  float nw = normw[t];
  const float* src = xtok + (size_t)t * CoutC + cg * 4;
  float* dst = outx + ((size_t)(b * Sc + cl)) * CoutC + cg * 4;
  #pragma unroll
  for (int c = 0; c < 4; ++c) atomicAdd(dst + c, src[c] * nw);
}

__global__ void k_final_a(const int* __restrict__ idxagg, const int* __restrict__ idxc,
                          const float* __restrict__ aggw, const float* __restrict__ normw,
                          float* __restrict__ out_idx, float* __restrict__ out_awd,
                          unsigned* __restrict__ awdmax){
  int i = blockIdx.x * blockDim.x + threadIdx.x;
  if (i >= Bn * Ntok) return;
  int b = i >> 12;
  int j = idxagg[i];
  int cl = idxc[b * Ntok + j];
  out_idx[i] = (float)cl;
  float awd = aggw[i] * normw[b * Ntok + j];
  out_awd[i] = awd;
  atomicMax(awdmax + b, __float_as_uint(awd));
}

__global__ void k_final_b(float* __restrict__ out_awd, const unsigned* __restrict__ awdmax){
  int i = blockIdx.x * blockDim.x + threadIdx.x;
  if (i >= Bn * Ntok) return;
  int b = i >> 12;
  out_awd[i] = out_awd[i] / __uint_as_float(awdmax[b]);
}

// =====================================================================================
extern "C" void kernel_launch(void* const* d_in, const int* in_sizes, int n_in,
                              void* d_out, int out_size, void* d_ws, size_t ws_size,
                              hipStream_t stream){
  (void)in_sizes; (void)n_in; (void)out_size; (void)ws_size;
  const float* x      = (const float*)d_in[0];
  const float* loc    = (const float*)d_in[1];
  const int*   idxagg = (const int*)d_in[2];
  const float* aggw   = (const float*)d_in[3];
  const float* convw  = (const float*)d_in[7];
  const float* convb  = (const float*)d_in[8];
  const float* skipw  = (const float*)d_in[9];
  const float* lng    = (const float*)d_in[10];
  const float* lnb    = (const float*)d_in[11];
  const float* confw  = (const float*)d_in[12];
  const float* confb  = (const float*)d_in[13];

  char* ws = (char*)d_ws;
  size_t off = 0;
  auto take = [&](size_t bytes) -> char* {
    char* p = ws + off;
    off += (bytes + 255) & ~(size_t)255;
    return p;
  };

  float* map_acc  = (float*)take((size_t)Bn * Hm * Wm * CinC * 4);
  float* map_cnt  = (float*)take((size_t)Bn * Hm * Wm * 4);
  u16t*  map_bf   = (u16t*) take((size_t)Bn * Hm * Wm * CinC * 2);
  u16t*  wtT      = (u16t*) take((size_t)CoutC * Kconv * 2);
  u16t*  skipT    = (u16t*) take((size_t)CoutC * CinC * 2);
  u16t*  x_bf     = (u16t*) take((size_t)Bn * Ntok * CinC * 2);
  float* conv_out = (float*)take((size_t)Bn * HoW * HoW * CoutC * 4);
  float* mp_acc   = (float*)take((size_t)Bn * Ntok * CoutC * 4);
  float* mp_wsum  = (float*)take((size_t)Bn * Ntok * 4);
  float* x_tok    = (float*)take((size_t)Bn * Ntok * CoutC * 4);
  u16t*  xb       = (u16t*) take((size_t)Bn * Ntok * CoutC * 2);
  float* x2       = (float*)take((size_t)Bn * Ntok * 4);
  float* wgt      = (float*)take((size_t)Bn * Ntok * 4);
  float* density  = (float*)take((size_t)Bn * Ntok * 4);
  float* score    = (float*)take((size_t)Bn * Ntok * 4);
  float* normw    = (float*)take((size_t)Bn * Ntok * 4);
  unsigned* dmaxu = (unsigned*)take(Bn * 4);
  unsigned* awdmx = (unsigned*)take(Bn * 4);
  int*   idx_down = (int*)  take((size_t)Bn * Sc * 4);
  int*   idx_clu  = (int*)  take((size_t)Bn * Ntok * 4);
  u16t*  xc       = (u16t*) take((size_t)Bn * Sc * CoutC * 2);
  float* x2c      = (float*)take((size_t)Bn * Sc * 4);
  float* allw     = (float*)take((size_t)Bn * Sc * 4);

  float* out_x   = (float*)d_out;                    // [Bn,Sc,CoutC]
  float* out_idx = out_x + (size_t)Bn * Sc * CoutC;  // [Bn,Ntok]
  float* out_awd = out_idx + (size_t)Bn * Ntok;      // [Bn,Ntok]

  // zero accumulators (graph-capture safe)
  hipMemsetAsync(map_acc, 0, (size_t)Bn * Hm * Wm * CinC * 4, stream);
  hipMemsetAsync(map_cnt, 0, (size_t)Bn * Hm * Wm * 4, stream);
  hipMemsetAsync(mp_acc,  0, (size_t)Bn * Ntok * CoutC * 4, stream);
  hipMemsetAsync(mp_wsum, 0, (size_t)Bn * Ntok * 4, stream);
  hipMemsetAsync(allw,    0, (size_t)Bn * Sc * 4, stream);
  hipMemsetAsync(dmaxu,   0, Bn * 4, stream);
  hipMemsetAsync(awdmx,   0, Bn * 4, stream);
  hipMemsetAsync(out_x,   0, (size_t)Bn * Sc * CoutC * 4, stream);

  const int TB = 256;
  auto gr = [&](long long n) -> unsigned { return (unsigned)((n + TB - 1) / TB); };

  // 1) token2map scatter + normalize to bf16
  k_scatter_map<<<gr((long long)Bn * Ntok * (CinC / 4)), TB, 0, stream>>>(x, loc, idxagg, map_acc, map_cnt);
  k_map_norm<<<gr((long long)Bn * Hm * Wm * CinC), TB, 0, stream>>>(map_acc, map_cnt, map_bf);

  // 2) weight preps
  k_transpose_bf16<<<gr((long long)Kconv * CoutC), TB, 0, stream>>>(convw, wtT, Kconv, CoutC);
  k_transpose_bf16<<<gr((long long)CinC * CoutC), TB, 0, stream>>>(skipw, skipT, CinC, CoutC);
  k_cast_bf16<<<gr((long long)Bn * Ntok * CinC), TB, 0, stream>>>(x, x_bf, Bn * Ntok * CinC);

  // 3) conv as implicit WMMA GEMM  (256 x 16 wave tiles of 16x32 = 4096 waves)
  k_conv_wmma<<<(4096 * 32) / TB, TB, 0, stream>>>(map_bf, wtT, convb, conv_out);

  // 4) map2token gather/scatter
  k_map2token<<<gr((long long)Bn * Ntok * (CoutC / 4)), TB, 0, stream>>>(conv_out, loc, idxagg, aggw, mp_acc, mp_wsum);

  // 5) skip GEMM: x_tok = x @ skip_w  (512 x 16 wave tiles of 32x32 = 8192 waves)
  k_wmma_gemm2<<<(8192 * 32) / TB, TB, 0, stream>>>(x_bf, skipT, x_tok, Bn * Ntok, CoutC, CinC);

  // 6) combine + layernorm + conf
  k_token_fuse<<<Bn * Ntok, 128, 0, stream>>>(mp_acc, mp_wsum, lng, lnb, confw, confb,
                                              x_tok, xb, x2, wgt);

  // 7) DPC-kNN: density pass (X X^T streaming), parent/score pass
  k_density<<<Bn * (Ntok / 32), 32, 0, stream>>>(xb, x2, density, dmaxu);
  k_parent_score<<<Bn * (Ntok / 32), 32, 0, stream>>>(xb, x2, density, dmaxu, score);

  // 8) top-S selection + center gather + assignment
  k_select<<<Bn, 1024, 0, stream>>>(score, idx_down);
  k_gather_centers<<<gr((long long)Bn * Sc * (CoutC / 4)), TB, 0, stream>>>(xb, x2, idx_down, xc, x2c);
  k_assign<<<Bn * (Ntok / 32), 32, 0, stream>>>(xc, xb, x2c, x2, idx_clu);

  // 9) merge tokens
  k_allw<<<gr((long long)Bn * Ntok), TB, 0, stream>>>(wgt, idx_clu, allw);
  k_normw<<<gr((long long)Bn * Ntok), TB, 0, stream>>>(wgt, allw, idx_clu, normw);
  k_xdown<<<gr((long long)Bn * Ntok * (CoutC / 4)), TB, 0, stream>>>(x_tok, idx_clu, normw, out_x);

  // 10) final index / weight outputs
  k_final_a<<<gr((long long)Bn * Ntok), TB, 0, stream>>>(idxagg, idx_clu, aggw, normw,
                                                         out_idx, out_awd, awdmx);
  k_final_b<<<gr((long long)Bn * Ntok), TB, 0, stream>>>(out_awd, awdmx);
}
